// VQVAE_10969346474534
// MI455X (gfx1250) — matrix operand order repaired and checked by reading, not compiled, not run on hardware
//
#include <hip/hip_runtime.h>

typedef __attribute__((ext_vector_type(16))) __bf16 v16bf;
typedef __attribute__((ext_vector_type(8)))  float  v8f;
typedef unsigned int v4u __attribute__((ext_vector_type(4)));
typedef int          v8i __attribute__((ext_vector_type(8)));
typedef int          v4i __attribute__((ext_vector_type(4)));

#if __has_builtin(__builtin_amdgcn_tensor_load_to_lds) && \
    __has_builtin(__builtin_amdgcn_s_wait_tensorcnt)
#define HAS_TDM 1
#endif

#define DEV static __device__ __forceinline__

DEV unsigned short f2bf(float f) {
  union { float f; unsigned u; } v; v.f = f;
  unsigned r = v.u + 0x7FFFu + ((v.u >> 16) & 1u);
  return (unsigned short)(r >> 16);
}

union FragU { v16bf v; uint4 q[2]; };

// A-fragment (16x32 bf16, MxK): lane&15 = M row; lanes 0-15 hold K 0..7,16..23,
// lanes 16-31 hold K 8..15,24..31 (ISA 7.12.2).
DEV v16bf load_fragA(const unsigned short* base, int half) {
  FragU f;
  f.q[0] = *(const uint4*)(base + half * 8);
  f.q[1] = *(const uint4*)(base + 16 + half * 8);
  return f.v;
}
// B-fragment (32x16 bf16, KxN) from an [n][k] (ld=32) LDS tile: lane&15 = N col,
// lanes 0-15 hold K 0..7,16..23, lanes 16-31 hold K 8..15,24..31 (same as A).
DEV v16bf load_fragB(const unsigned short* rowbase, int half) {
  FragU f;
  f.q[0] = *(const uint4*)(rowbase + half * 8);
  f.q[1] = *(const uint4*)(rowbase + 16 + half * 8);
  return f.v;
}

#ifdef HAS_TDM
// Issue a TDM load of a fully-in-bounds 2D bf16 tile (tile_w elems per row,
// tile_h rows, row stride in elems) from global into LDS at lds_addr.
// D# packing per CDNA5 ISA 8.3/8.4: count=1, type=2(image), data_size=1(2B).
DEV void tdm_load_2d_bf16(unsigned lds_addr, const unsigned short* gaddr,
                          int tile_w, int tile_h, long long row_stride_elems) {
  unsigned long long ga = (unsigned long long)(size_t)gaddr;
  unsigned long long st = (unsigned long long)row_stride_elems;
  unsigned td0 = (unsigned)tile_w, td1 = (unsigned)tile_h;
  v4u g0;
  g0[0] = 1u;                                            // count=1, user mode
  g0[1] = lds_addr;                                      // [63:32] lds_addr
  g0[2] = (unsigned)ga;                                  // global_addr[31:0]
  g0[3] = (unsigned)((ga >> 32) & 0x1FFFFFFu) | (2u << 30); // addr[56:32] | type=2
  v8i g1;
  g1[0] = (int)(1u << 16);                               // data_size=1 (2 bytes)
  g1[1] = (int)((td0 & 0xFFFFu) << 16);                  // tensor_dim0[15:0] @48
  g1[2] = (int)(((td0 >> 16) & 0xFFFFu) | ((td1 & 0xFFFFu) << 16)); // dim0 hi | dim1 lo
  g1[3] = (int)(((td1 >> 16) & 0xFFFFu) | ((td0 & 0xFFFFu) << 16)); // dim1 hi | tile_dim0
  g1[4] = (int)(td1 & 0xFFFFu);                          // tile_dim1 | tile_dim2=0
  g1[5] = (int)(st & 0xFFFFFFFFu);                       // tensor_dim0_stride lo
  g1[6] = (int)((st >> 32) & 0xFFFFu);                   // stride hi | dim1_stride lo=0
  g1[7] = 0;
  v4i z4 = {0, 0, 0, 0};
#if defined(__clang_major__) && (__clang_major__ >= 23)
  v8i z8 = {0, 0, 0, 0, 0, 0, 0, 0};
  __builtin_amdgcn_tensor_load_to_lds(g0, g1, z4, z4, z8, 0);
#else
  __builtin_amdgcn_tensor_load_to_lds(g0, g1, z4, z4, 0);
#endif
}
#endif

struct ConvP {
  int B, Cin, H, W, Cout, OH, OW, KH, KW, stride, pad, convt, Ktot;
};

// ---------------- implicit-GEMM conv / convT via WMMA bf16 ----------------
// Out[b,co,oh,ow] = sum_k Wp[co,k] * patch(k; b,oh,ow)
// Workgroup tile: M=128 (Cout), N=64 (pixels). 8 waves as 4x2, each 32x32.
// A-tile (weights) arrives via the Tensor Data Mover; B-tile is an im2col
// gather done by all waves in parallel with the DMA.
__global__ __launch_bounds__(256)
void conv_wmma_kernel(const unsigned short* __restrict__ Wp,  // [Cout,Ktot] bf16
                      const unsigned short* __restrict__ Xb,  // [B,Cin,H,W] bf16
                      float* __restrict__ Out, ConvP P) {
  __shared__ unsigned short As[128 * 32];  // [m][k]
  __shared__ unsigned short Bs[64 * 32];   // [n][k]
  const int tid = threadIdx.x;
  const int lane = tid & 31, wave = tid >> 5;
  const int wm = wave >> 1, wn = wave & 1;
  const int blockN = blockIdx.x * 64;
  const int blockM = blockIdx.y * 128;
  const int OHW = P.OH * P.OW;
  const int KHW = P.KH * P.KW;
  v8f acc[4] = {};

  for (int k0 = 0; k0 < P.Ktot; k0 += 32) {
#ifdef HAS_TDM
    if (wave == 0)  // one wave drives the DMA of the 128x32 weight tile
      tdm_load_2d_bf16((unsigned)(size_t)As,
                       Wp + (size_t)blockM * P.Ktot + k0, 32, 128, P.Ktot);
#else
    {  // A: 128x32 bf16, 2 threads per row
      int r = tid >> 1;
      int hk = (tid & 1) * 16;
      const unsigned short* src = Wp + (size_t)(blockM + r) * P.Ktot + k0 + hk;
      *(uint4*)&As[r * 32 + hk]     = *(const uint4*)(src);
      *(uint4*)&As[r * 32 + hk + 8] = *(const uint4*)(src + 8);
    }
#endif
    {  // B: gather 64 pixels x 32 k (implicit im2col), incremental k-decompose
      int nl = tid >> 2;
      int kq = (tid & 3) * 8;
      int p = blockN + nl;
      int b = p / OHW, s = p - b * OHW;
      int oh = s / P.OW, ow = s - oh * P.OW;
      int kg = k0 + kq;
      int ci = kg / KHW; int rem = kg - ci * KHW;
      int kh = rem / P.KW, kw = rem - kh * P.KW;
      for (int j = 0; j < 8; ++j) {
        unsigned short v = 0;
        if (P.convt) {
          int dh = oh - P.pad + kh, dw = ow - P.pad + kw;
          if (dh >= 0 && dw >= 0 && !(dh & 1) && !(dw & 1)) {
            int ih = dh >> 1, iw = dw >> 1;
            if (ih < P.H && iw < P.W)
              v = Xb[(((size_t)b * P.Cin + ci) * P.H + ih) * P.W + iw];
          }
        } else {
          int ih = oh * P.stride - P.pad + kh;
          int iw = ow * P.stride - P.pad + kw;
          if (ih >= 0 && iw >= 0 && ih < P.H && iw < P.W)
            v = Xb[(((size_t)b * P.Cin + ci) * P.H + ih) * P.W + iw];
        }
        Bs[nl * 32 + kq + j] = v;
        if (++kw == P.KW) { kw = 0; if (++kh == P.KH) { kh = 0; ++ci; } }
      }
    }
#ifdef HAS_TDM
    if (wave == 0) __builtin_amdgcn_s_wait_tensorcnt(0);
#endif
    __syncthreads();
    const int half = lane >> 4;
    for (int mi = 0; mi < 2; ++mi) {
      v16bf a = load_fragA(&As[(wm * 32 + mi * 16 + (lane & 15)) * 32], half);
      for (int ni = 0; ni < 2; ++ni) {
        v16bf b = load_fragB(&Bs[(wn * 32 + ni * 16 + (lane & 15)) * 32], half);
        acc[mi * 2 + ni] = __builtin_amdgcn_wmma_f32_16x16x32_bf16(
            false, a, false, b, (short)0, acc[mi * 2 + ni], false, false);
      }
    }
    __syncthreads();
  }
  // C layout (ISA 7.12.2): N = lane&15, M = r + 8*(lane>>4)
  for (int mi = 0; mi < 2; ++mi)
    for (int ni = 0; ni < 2; ++ni) {
      v8f c = acc[mi * 2 + ni];
      int pcol = blockN + wn * 32 + ni * 16 + (lane & 15);
      int b = pcol / OHW, s = pcol - b * OHW;
      for (int r = 0; r < 8; ++r) {
        int co = blockM + wm * 32 + mi * 16 + r + 8 * (lane >> 4);
        Out[((size_t)b * P.Cout + co) * OHW + s] = c[r];
      }
    }
}

// ---------------- VQ: argmin_k ||x_n||^2 - 2 x_n.c_k + ||c_k||^2 -----------
__global__ __launch_bounds__(256)
void vq_argmin_kernel(const unsigned short* __restrict__ Flat,  // [N,256] bf16
                      const unsigned short* __restrict__ Cb,    // [512,256] bf16
                      const float* __restrict__ Cnorm,          // [512]
                      int* __restrict__ Idx) {
  __shared__ unsigned short Af[64 * 256];
  __shared__ unsigned short Bs[64 * 32];
  __shared__ float Gs[64 * 68];
  __shared__ float RBv[256];
  __shared__ int   RBi[256];
  const int tid = threadIdx.x, lane = tid & 31, wave = tid >> 5;
  const int wm = wave >> 1, wn = wave & 1;
  const int row0 = blockIdx.x * 64;

#ifdef HAS_TDM
  if (wave == 0) {  // 64x256 bf16 point block is one contiguous 32KB strip
    tdm_load_2d_bf16((unsigned)(size_t)Af, Flat + (size_t)row0 * 256,
                     16384, 1, 16384);
    __builtin_amdgcn_s_wait_tensorcnt(0);
  }
#else
  for (int i = tid; i < (64 * 256) / 8; i += 256)
    ((uint4*)Af)[i] = ((const uint4*)(Flat + (size_t)row0 * 256))[i];
#endif
  __syncthreads();

  float best = 3.4e38f; int bidx = 0;
  const int myrow = tid >> 2, myq = tid & 3;

  for (int nc0 = 0; nc0 < 512; nc0 += 64) {
    v8f acc[2] = {};
    for (int k0 = 0; k0 < 256; k0 += 32) {
#ifdef HAS_TDM
      if (wave == 0) {  // 64 codebook rows x 32 elems, row stride 256
        tdm_load_2d_bf16((unsigned)(size_t)Bs,
                         Cb + (size_t)nc0 * 256 + k0, 32, 64, 256);
        __builtin_amdgcn_s_wait_tensorcnt(0);
      }
#else
      {
        int nl = tid >> 2, kq = (tid & 3) * 8;
        *(uint4*)&Bs[nl * 32 + kq] =
            *(const uint4*)(Cb + (size_t)(nc0 + nl) * 256 + k0 + kq);
      }
#endif
      __syncthreads();
      const int half = lane >> 4;
      v16bf a = load_fragA(&Af[(wm * 16 + (lane & 15)) * 256 + k0], half);
      for (int ni = 0; ni < 2; ++ni) {
        v16bf b = load_fragB(&Bs[(wn * 32 + ni * 16 + (lane & 15)) * 32], half);
        acc[ni] = __builtin_amdgcn_wmma_f32_16x16x32_bf16(
            false, a, false, b, (short)0, acc[ni], false, false);
      }
      __syncthreads();
    }
    for (int ni = 0; ni < 2; ++ni) {
      int col = wn * 32 + ni * 16 + (lane & 15);
      for (int r = 0; r < 8; ++r) {
        int m = wm * 16 + r + 8 * (lane >> 4);
        Gs[m * 68 + col] = acc[ni][r];
      }
    }
    __syncthreads();
    for (int j = 0; j < 16; ++j) {
      int c = myq * 16 + j;
      float d = Cnorm[nc0 + c] - 2.0f * Gs[myrow * 68 + c];
      if (d < best) { best = d; bidx = nc0 + c; }
    }
    __syncthreads();
  }
  RBv[tid] = best; RBi[tid] = bidx;
  __syncthreads();
  if (myq == 0) {
    float bv = RBv[tid]; int bi = RBi[tid];
    for (int q = 1; q < 4; ++q) {
      float v = RBv[tid + q]; int i2 = RBi[tid + q];
      if (v < bv || (v == bv && i2 < bi)) { bv = v; bi = i2; }
    }
    Idx[row0 + myrow] = bi;
  }
}

// ---------------- BatchNorm (training-mode batch stats) -------------------
__global__ __launch_bounds__(256)
void bn_stats_kernel(const float* __restrict__ X, float* __restrict__ stats,
                     int C, int spb, int Bn, float eps) {
  __shared__ float r1[256], r2[256];
  const int c = blockIdx.x, t = threadIdx.x;
  float s = 0.f, s2 = 0.f;
  const int n = Bn * spb;
  for (int i = t; i < n; i += 256) {
    int b = i / spb, r = i - b * spb;
    float v = X[((size_t)b * C + c) * spb + r];
    s += v; s2 += v * v;
  }
  r1[t] = s; r2[t] = s2;
  __syncthreads();
  for (int k = 128; k > 0; k >>= 1) {
    if (t < k) { r1[t] += r1[t + k]; r2[t] += r2[t + k]; }
    __syncthreads();
  }
  if (t == 0) {
    float cnt = (float)n;
    float m = r1[0] / cnt;
    float var = r2[0] / cnt - m * m;
    stats[c] = m;
    stats[C + c] = rsqrtf(var + eps);
  }
}

// mode: 0 = none, 1 = relu, 2 = tanh. OutF / OutB may be null.
__global__ __launch_bounds__(256)
void bn_apply_kernel(const float* __restrict__ X, const float* __restrict__ stats,
                     const float* __restrict__ g, const float* __restrict__ bta,
                     float* __restrict__ OutF, unsigned short* __restrict__ OutB,
                     int C, int spb, size_t total, int mode) {
  size_t i = (size_t)blockIdx.x * blockDim.x + threadIdx.x;
  size_t st = (size_t)gridDim.x * blockDim.x;
  for (; i < total; i += st) {
    int c = (int)((i / spb) % C);
    float y = g[c] * (X[i] - stats[c]) * stats[C + c] + bta[c];
    if (mode == 1) y = fmaxf(y, 0.f);
    else if (mode == 2) y = tanhf(y);
    if (OutF) OutF[i] = y;
    if (OutB) OutB[i] = f2bf(y);
  }
}

// ---------------- small elementwise / pack kernels ------------------------
__global__ __launch_bounds__(256)
void k_f32_to_bf16(const float* __restrict__ x, unsigned short* __restrict__ y, size_t n) {
  size_t i = (size_t)blockIdx.x * blockDim.x + threadIdx.x;
  size_t st = (size_t)gridDim.x * blockDim.x;
  for (; i < n; i += st) y[i] = f2bf(x[i]);
}

__global__ __launch_bounds__(256)
void k_add(const float* __restrict__ a, const float* __restrict__ b,
           float* __restrict__ c, size_t n) {
  size_t i = (size_t)blockIdx.x * blockDim.x + threadIdx.x;
  size_t st = (size_t)gridDim.x * blockDim.x;
  for (; i < n; i += st) c[i] = a[i] + b[i];
}

// d_t1 (in,out,kh,kw) -> flipped+transposed [co][ci][kh][kw] bf16
__global__ __launch_bounds__(256)
void k_pack_wt1(const float* __restrict__ W, unsigned short* __restrict__ Y) {
  size_t i = (size_t)blockIdx.x * blockDim.x + threadIdx.x;
  size_t st = (size_t)gridDim.x * blockDim.x;
  const size_t n = 256u * 256u * 16u;
  for (; i < n; i += st) {
    int co = (int)(i / (256 * 16));
    int rem = (int)(i - (size_t)co * 256 * 16);
    int ci = rem >> 4, k = rem & 15;
    int kh = k >> 2, kw = k & 3;
    Y[i] = f2bf(W[(((size_t)ci * 256 + co) * 4 + (3 - kh)) * 4 + (3 - kw)]);
  }
}

__global__ __launch_bounds__(256)
void k_pack_codebook(const float* __restrict__ Cb, unsigned short* __restrict__ CbB,
                     float* __restrict__ Cnorm) {
  __shared__ float red[256];
  const int code = blockIdx.x, t = threadIdx.x;
  float v = Cb[(size_t)code * 256 + t];
  CbB[(size_t)code * 256 + t] = f2bf(v);
  red[t] = v * v;
  __syncthreads();
  for (int s = 128; s > 0; s >>= 1) {
    if (t < s) red[t] += red[t + s];
    __syncthreads();
  }
  if (t == 0) Cnorm[code] = red[0];
}

// z_e NCHW f32 -> flat [N, C] bf16
__global__ __launch_bounds__(256)
void k_flatten(const float* __restrict__ Z, unsigned short* __restrict__ F,
               int C, int spb, size_t total) {
  size_t i = (size_t)blockIdx.x * blockDim.x + threadIdx.x;
  size_t st = (size_t)gridDim.x * blockDim.x;
  for (; i < total; i += st) {
    size_t n = i / C; int c = (int)(i - n * C);
    size_t b = n / spb, s = n - b * spb;
    F[i] = f2bf(Z[((size_t)b * C + c) * spb + s]);
  }
}

__global__ __launch_bounds__(256)
void k_vq_gather(const int* __restrict__ Idx, const float* __restrict__ Cb,
                 float* __restrict__ Zq, unsigned short* __restrict__ ZqB,
                 int C, int spb, size_t total) {
  size_t i = (size_t)blockIdx.x * blockDim.x + threadIdx.x;
  size_t st = (size_t)gridDim.x * blockDim.x;
  for (; i < total; i += st) {
    size_t n = i / C; int c = (int)(i - n * C);
    size_t b = n / spb, s = n - b * spb;
    float v = Cb[(size_t)Idx[n] * C + c];
    size_t o = ((size_t)b * C + c) * spb + s;
    Zq[o] = v; ZqB[o] = f2bf(v);
  }
}

// ---------------- direct VALU convs for channel-3 ends --------------------
// conv1: 3->256, k4 s2 p1 on 256x256
__global__ __launch_bounds__(256)
void conv1_direct(const float* __restrict__ X, const float* __restrict__ W1,
                  float* __restrict__ Y) {
  const int b = blockIdx.x >> 8, co = blockIdx.x & 255;
  __shared__ float ws[48];
  if (threadIdx.x < 48) ws[threadIdx.x] = W1[co * 48 + threadIdx.x];
  __syncthreads();
  for (int i = threadIdx.x; i < 128 * 128; i += 256) {
    int oh = i >> 7, ow = i & 127;
    float acc = 0.f;
    for (int ci = 0; ci < 3; ++ci)
      for (int kh = 0; kh < 4; ++kh) {
        int ih = oh * 2 - 1 + kh;
        if ((unsigned)ih >= 256u) continue;
        for (int kw = 0; kw < 4; ++kw) {
          int iw = ow * 2 - 1 + kw;
          if ((unsigned)iw >= 256u) continue;
          acc += ws[(ci * 4 + kh) * 4 + kw] *
                 X[(((size_t)b * 3 + ci) * 256 + ih) * 256 + iw];
        }
      }
    Y[(((size_t)b * 256 + co) * 128 + oh) * 128 + ow] = acc;
  }
}

// convT2: 256->3, k4 s2 p1, 128x128 -> 256x256
__global__ __launch_bounds__(256)
void convt2_direct(const float* __restrict__ X, const float* __restrict__ Wt,
                   float* __restrict__ Y) {
  const int bc = blockIdx.x >> 6;    // (b,co)
  const int chunk = blockIdx.x & 63;
  const int b = bc / 3, co = bc - b * 3;
  __shared__ float ws[256 * 16];
  for (int i = threadIdx.x; i < 256 * 16; i += 256) {
    int ci = i >> 4, kh = (i >> 2) & 3, kw = i & 3;
    ws[i] = Wt[(((size_t)ci * 3 + co) * 4 + (3 - kh)) * 4 + (3 - kw)];
  }
  __syncthreads();
  for (int t = threadIdx.x; t < 1024; t += 256) {
    int o = chunk * 1024 + t;
    int oh = o >> 8, ow = o & 255;
    float acc = 0.f;
    for (int kh = 0; kh < 4; ++kh) {
      int dh = oh - 2 + kh;
      if (dh < 0 || (dh & 1)) continue;
      int ih = dh >> 1; if (ih >= 128) continue;
      for (int kw = 0; kw < 4; ++kw) {
        int dw = ow - 2 + kw;
        if (dw < 0 || (dw & 1)) continue;
        int iw = dw >> 1; if (iw >= 128) continue;
        const float* xp = X + (((size_t)b * 256) * 128 + ih) * 128 + iw;
        for (int ci = 0; ci < 256; ++ci)
          acc += ws[(ci << 4) + (kh << 2) + kw] * xp[(size_t)ci * 16384];
      }
    }
    Y[(((size_t)b * 3 + co) * 256 + oh) * 256 + ow] = acc;
  }
}

// ======================= host-side orchestration ==========================
extern "C" void kernel_launch(void* const* d_in, const int* in_sizes, int n_in,
                              void* d_out, int out_size, void* d_ws, size_t ws_size,
                              hipStream_t stream) {
  const float EPS = 1e-5f;
  const float* ori  = (const float*)d_in[0];
  const float* e_w1 = (const float*)d_in[1];
  const float* e_g1 = (const float*)d_in[2];  const float* e_b1 = (const float*)d_in[3];
  const float* e_w2 = (const float*)d_in[4];
  const float* e_g2 = (const float*)d_in[5];  const float* e_b2 = (const float*)d_in[6];
  const float* e_w3 = (const float*)d_in[7];
  const float* e_g3 = (const float*)d_in[8];  const float* e_b3 = (const float*)d_in[9];
  const float* e_w4 = (const float*)d_in[10]; /* e_c4 = d_in[11] cancels under BN */
  const float* e_g4 = (const float*)d_in[12]; const float* e_b4 = (const float*)d_in[13];
  const float* e_g5 = (const float*)d_in[14]; const float* e_b5 = (const float*)d_in[15];
  const float* codebook = (const float*)d_in[16];
  const float* d_w1 = (const float*)d_in[17];
  const float* d_g1 = (const float*)d_in[18]; const float* d_b1 = (const float*)d_in[19];
  const float* d_w2 = (const float*)d_in[20];
  const float* d_g2 = (const float*)d_in[21]; const float* d_b2 = (const float*)d_in[22];
  const float* d_g3 = (const float*)d_in[23]; const float* d_b3 = (const float*)d_in[24];
  const float* d_t1 = (const float*)d_in[25];
  const float* d_g4 = (const float*)d_in[26]; const float* d_b4 = (const float*)d_in[27];
  const float* d_t2 = (const float*)d_in[28]; /* d_c2 = d_in[29] cancels under BN */
  const float* d_g5 = (const float*)d_in[30]; const float* d_b5 = (const float*)d_in[31];
  float* out = (float*)d_out;

  const size_t E128 = 33554432;  // 8*256*128*128
  const size_t E64  = 8388608;   // 8*256*64*64
  const size_t NPTS = 32768;     // 8*64*64

  char* base = (char*)d_ws; size_t off = 0;
  auto alloc = [&](size_t bytes) -> void* {
    void* p = base + off;
    off += (bytes + 255) & ~(size_t)255;
    return p;
  };
  float*          BIGF0 = (float*)alloc(E128 * 4);
  unsigned short* BIGB0 = (unsigned short*)alloc(E128 * 2);
  float*          M0 = (float*)alloc(E64 * 4);
  float*          M1 = (float*)alloc(E64 * 4);
  float*          M2 = (float*)alloc(E64 * 4);
  unsigned short* MB0 = (unsigned short*)alloc(E64 * 2);
  unsigned short* MB1 = (unsigned short*)alloc(E64 * 2);
  unsigned short* FB  = (unsigned short*)alloc(NPTS * 256 * 2);
  unsigned short* CBB = (unsigned short*)alloc(512 * 256 * 2);
  float*          CN  = (float*)alloc(512 * 4);
  int*            IDX = (int*)alloc(NPTS * 4);
  float*          ST  = (float*)alloc(512 * 4);
  unsigned short* Wp2 = (unsigned short*)alloc(256 * 4096 * 2);
  unsigned short* Wp3 = (unsigned short*)alloc(256 * 2304 * 2);
  unsigned short* Wp4 = (unsigned short*)alloc(256 * 256 * 2);
  unsigned short* Wd1 = (unsigned short*)alloc(256 * 2304 * 2);
  unsigned short* Wd2 = (unsigned short*)alloc(256 * 256 * 2);
  unsigned short* Wt1 = (unsigned short*)alloc(256 * 4096 * 2);

  const int GS = 2048;  // grid-stride block count
#define EW(kern, ...) kern<<<GS, 256, 0, stream>>>(__VA_ARGS__)

  // ---- weight / codebook packing ----
  EW(k_f32_to_bf16, e_w2, Wp2, (size_t)256 * 4096);
  EW(k_f32_to_bf16, e_w3, Wp3, (size_t)256 * 2304);
  EW(k_f32_to_bf16, e_w4, Wp4, (size_t)256 * 256);
  EW(k_f32_to_bf16, d_w1, Wd1, (size_t)256 * 2304);
  EW(k_f32_to_bf16, d_w2, Wd2, (size_t)256 * 256);
  EW(k_pack_wt1, d_t1, Wt1);
  k_pack_codebook<<<512, 256, 0, stream>>>(codebook, CBB, CN);

  // ---- encoder ----
  conv1_direct<<<8 * 256, 256, 0, stream>>>(ori, e_w1, BIGF0);
  bn_stats_kernel<<<256, 256, 0, stream>>>(BIGF0, ST, 256, 16384, 8, EPS);
  EW(bn_apply_kernel, BIGF0, ST, e_g1, e_b1, (float*)nullptr, BIGB0, 256, 16384, E128, 1);

  ConvP pc2 = {8, 256, 128, 128, 256, 64, 64, 4, 4, 2, 1, 0, 4096};
  conv_wmma_kernel<<<dim3(512, 2), 256, 0, stream>>>(Wp2, BIGB0, M0, pc2);
  bn_stats_kernel<<<256, 256, 0, stream>>>(M0, ST, 256, 4096, 8, EPS);
  EW(bn_apply_kernel, M0, ST, e_g2, e_b2, M1, MB0, 256, 4096, E64, 1);  // x2

  ConvP pc3 = {8, 256, 64, 64, 256, 64, 64, 3, 3, 1, 1, 0, 2304};
  conv_wmma_kernel<<<dim3(512, 2), 256, 0, stream>>>(Wp3, MB0, M2, pc3);
  bn_stats_kernel<<<256, 256, 0, stream>>>(M2, ST, 256, 4096, 8, EPS);
  EW(bn_apply_kernel, M2, ST, e_g3, e_b3, (float*)nullptr, MB1, 256, 4096, E64, 1);  // r3

  ConvP pc4 = {8, 256, 64, 64, 256, 64, 64, 1, 1, 1, 0, 0, 256};
  conv_wmma_kernel<<<dim3(512, 2), 256, 0, stream>>>(Wp4, MB1, M2, pc4);
  bn_stats_kernel<<<256, 256, 0, stream>>>(M2, ST, 256, 4096, 8, EPS);
  EW(bn_apply_kernel, M2, ST, e_g4, e_b4, M0, (unsigned short*)nullptr, 256, 4096, E64, 1);  // r4

  EW(k_add, M1, M0, M2, E64);  // s5 = x2 + r4
  bn_stats_kernel<<<256, 256, 0, stream>>>(M2, ST, 256, 4096, 8, EPS);
  EW(bn_apply_kernel, M2, ST, e_g5, e_b5, M0, (unsigned short*)nullptr, 256, 4096, E64, 0);  // z_e

  // ---- VQ ----
  EW(k_flatten, M0, FB, 256, 4096, E64);
  vq_argmin_kernel<<<512, 256, 0, stream>>>(FB, CBB, CN, IDX);
  EW(k_vq_gather, IDX, codebook, M1, MB0, 256, 4096, E64);  // z_q f32 + bf16

  // ---- decoder ----
  conv_wmma_kernel<<<dim3(512, 2), 256, 0, stream>>>(Wd1, MB0, M2, pc3);
  bn_stats_kernel<<<256, 256, 0, stream>>>(M2, ST, 256, 4096, 8, EPS);
  EW(bn_apply_kernel, M2, ST, d_g1, d_b1, (float*)nullptr, MB1, 256, 4096, E64, 1);

  conv_wmma_kernel<<<dim3(512, 2), 256, 0, stream>>>(Wd2, MB1, M0, pc4);
  bn_stats_kernel<<<256, 256, 0, stream>>>(M0, ST, 256, 4096, 8, EPS);
  EW(bn_apply_kernel, M0, ST, d_g2, d_b2, M2, (unsigned short*)nullptr, 256, 4096, E64, 1);

  EW(k_add, M1, M2, M0, E64);  // z_q + r
  bn_stats_kernel<<<256, 256, 0, stream>>>(M0, ST, 256, 4096, 8, EPS);
  EW(bn_apply_kernel, M0, ST, d_g3, d_b3, (float*)nullptr, MB1, 256, 4096, E64, 1);

  ConvP pt1 = {8, 256, 64, 64, 256, 128, 128, 4, 4, 1, 2, 1, 4096};  // convT k4 s2 p1
  conv_wmma_kernel<<<dim3(2048, 2), 256, 0, stream>>>(Wt1, MB1, BIGF0, pt1);
  bn_stats_kernel<<<256, 256, 0, stream>>>(BIGF0, ST, 256, 16384, 8, EPS);
  EW(bn_apply_kernel, BIGF0, ST, d_g4, d_b4, BIGF0, (unsigned short*)nullptr, 256, 16384, E128, 1);

  convt2_direct<<<8 * 3 * 64, 256, 0, stream>>>(BIGF0, d_t2, M2);
  bn_stats_kernel<<<3, 256, 0, stream>>>(M2, ST, 3, 65536, 8, EPS);
  EW(bn_apply_kernel, M2, ST, d_g5, d_b5, out, (unsigned short*)nullptr, 3, 65536,
     (size_t)8 * 3 * 256 * 256, 2);
#undef EW
  (void)in_sizes; (void)n_in; (void)out_size; (void)ws_size;
}
